// LNNFirstTransformerModel_61538291417789
// MI455X (gfx1250) — matrix operand order repaired and checked by reading, not compiled, not run on hardware
//
#include <hip/hip_runtime.h>
#include <math.h>

// ---------------------------------------------------------------------------
// LNN-first transformer for MI455X (gfx1250, wave32).
// - All dense math: v_wmma_f32_16x16x32_bf16 (f32 accumulate).
// - All activations + converted weights stored bf16 in workspace (halves
//   traffic, kills f32->bf16 converts in hot loops; ~100MB => L2-resident).
// - GEMM/attention tile staging uses GLOBAL_LOAD_ASYNC_TO_LDS_B128
//   (ASYNCcnt-tracked) via inline asm, per CDNA5 ISA.
// ---------------------------------------------------------------------------

typedef __bf16 bf16;
typedef __attribute__((ext_vector_type(16))) __bf16 v16bf;
typedef __attribute__((ext_vector_type(8)))  float  v8f;

#define WMMA_BF16(a, b, c) \
  __builtin_amdgcn_wmma_f32_16x16x32_bf16(false, (a), false, (b), (short)0, (c), false, false)

__device__ __forceinline__ float bperm(float x, int srclane) {
  return __int_as_float(__builtin_amdgcn_ds_bpermute(srclane << 2, __float_as_int(x)));
}
__device__ __forceinline__ float gelu_erf(float v) {
  return 0.5f * v * (1.0f + erff(v * 0.70710678118654752f));
}

// Async 16B memory->LDS copy (GV mode). Generic shared-pointer low 32 bits are
// the LDS byte offset (ISA aperture rule).
__device__ __forceinline__ void async_copy16(void* lds_dst, const void* gsrc) {
  uint32_t lds_off = (uint32_t)(uintptr_t)lds_dst;
  asm volatile("global_load_async_to_lds_b128 %0, %1, off"
               :: "v"(lds_off), "v"(gsrc) : "memory");
}
__device__ __forceinline__ void wait_async0() {
  asm volatile("s_wait_asynccnt 0x0" ::: "memory");
}

// ---------------------------------------------------------------------------
// One-time f32 -> bf16 conversion (weights).
// ---------------------------------------------------------------------------
__global__ void __launch_bounds__(256)
f32_to_bf16(const float* __restrict__ s, bf16* __restrict__ d, int n) {
  int i = blockIdx.x * 256 + threadIdx.x;
  if (i < n) d[i] = (bf16)s[i];
}

// ---------------------------------------------------------------------------
// GEMM: C(M,N) = act(A(M,K) @ W(N,K)^T + bias(N)); A,W bf16, C bf16, bias f32.
// 256 threads = 8 waves; tile 128(M) x 64(N); K staged 32-wide via async LDS.
// ---------------------------------------------------------------------------
template <int ACT>
__global__ void __launch_bounds__(256)
gemm_bias_act(const bf16* __restrict__ A, const bf16* __restrict__ W,
              const float* __restrict__ bias, bf16* __restrict__ C,
              int M, int N, int K) {
  __shared__ bf16 sA[128 * 40];  // stride 40 halves = 80B (16B aligned, bank-shifted)
  __shared__ bf16 sB[64 * 40];
  const int tid  = threadIdx.x;
  const int wave = tid >> 5;
  const int lane = tid & 31;
  const int hi   = lane >> 4;
  const int ln   = lane & 15;
  const int mbase = blockIdx.y * 128;
  const int nbase = blockIdx.x * 64;

  v8f acc[4];
#pragma unroll
  for (int i = 0; i < 4; ++i) acc[i] = (v8f)(0.0f);

  for (int kc = 0; kc < K; kc += 32) {
    // A tile: 128 rows x 32 halves = 512 x 16B chunks; 2 per thread.
#pragma unroll
    for (int it = 0; it < 2; ++it) {
      int cid = it * 256 + tid;
      int r = cid >> 2, sg = cid & 3;  // 4 x 16B per row
      async_copy16(&sA[r * 40 + sg * 8],
                   A + (size_t)(mbase + r) * K + kc + sg * 8);
    }
    // W tile: 64 rows x 32 halves = 256 chunks; 1 per thread.
    {
      int r = tid >> 2, sg = tid & 3;
      async_copy16(&sB[r * 40 + sg * 8],
                   W + (size_t)(nbase + r) * K + kc + sg * 8);
    }
    wait_async0();
    __syncthreads();

    // A fragment: lane = row, one contiguous 16-half K run (k = 16*hi + j)
    v16bf a;
    const bf16* ap = &sA[(wave * 16 + ln) * 40 + 16 * hi];
#pragma unroll
    for (int j = 0; j < 8; ++j) a[j] = ap[j];
#pragma unroll
    for (int j = 0; j < 8; ++j) a[8 + j] = ap[8 + j];

#pragma unroll
    for (int nt = 0; nt < 4; ++nt) {
      v16bf b;
      const bf16* bp = &sB[(nt * 16 + ln) * 40 + 16 * hi];
#pragma unroll
      for (int e = 0; e < 16; ++e) b[e] = bp[e];
      acc[nt] = WMMA_BF16(a, b, acc[nt]);
    }
    __syncthreads();
  }

#pragma unroll
  for (int nt = 0; nt < 4; ++nt) {
    int col = nbase + nt * 16 + ln;
    float bv = bias[col];
#pragma unroll
    for (int v = 0; v < 8; ++v) {
      int row = mbase + wave * 16 + v + 8 * hi;
      float val = acc[nt][v] + bv;
      if (ACT == 1) val = gelu_erf(val);
      C[(size_t)row * N + col] = (bf16)val;
    }
  }
}

// ---------------------------------------------------------------------------
// Input projection: u = gelu(LN(x * W_in^T + b_in)).  One block per (b,t) row.
// ---------------------------------------------------------------------------
__global__ void __launch_bounds__(256)
input_proj_ln_gelu(const float* __restrict__ x, const float* __restrict__ Win,
                   const float* __restrict__ bin, const float* __restrict__ g,
                   const float* __restrict__ be, bf16* __restrict__ u) {
  __shared__ float sb[256];
  const int row = blockIdx.x;
  const int h   = threadIdx.x;
  float v = x[row] * Win[h] + bin[h];
  sb[h] = v;
  __syncthreads();
  for (int s = 128; s > 0; s >>= 1) { if (h < s) sb[h] += sb[h + s]; __syncthreads(); }
  float mean = sb[0] * (1.0f / 256.0f);
  __syncthreads();
  float d = v - mean;
  sb[h] = d * d;
  __syncthreads();
  for (int s = 128; s > 0; s >>= 1) { if (h < s) sb[h] += sb[h + s]; __syncthreads(); }
  float var = sb[0] * (1.0f / 256.0f);
  float y = d * rsqrtf(var + 1e-5f) * g[h] + be[h];
  u[(size_t)row * 256 + h] = (bf16)gelu_erf(y);
}

// ---------------------------------------------------------------------------
// seq = LN(res + y) per row (safe in-place on res: one element per thread).
// ---------------------------------------------------------------------------
__global__ void __launch_bounds__(256)
residual_ln(const bf16* res, const bf16* y,
            const float* __restrict__ g, const float* __restrict__ be,
            bf16* out) {
  __shared__ float sb[256];
  const int row = blockIdx.x;
  const int h   = threadIdx.x;
  float v = (float)res[(size_t)row * 256 + h] + (float)y[(size_t)row * 256 + h];
  sb[h] = v;
  __syncthreads();
  for (int s = 128; s > 0; s >>= 1) { if (h < s) sb[h] += sb[h + s]; __syncthreads(); }
  float mean = sb[0] * (1.0f / 256.0f);
  __syncthreads();
  float d = v - mean;
  sb[h] = d * d;
  __syncthreads();
  for (int s = 128; s > 0; s >>= 1) { if (h < s) sb[h] += sb[h + s]; __syncthreads(); }
  float var = sb[0] * (1.0f / 256.0f);
  out[(size_t)row * 256 + h] = (bf16)(d * rsqrtf(var + 1e-5f) * g[h] + be[h]);
}

// ---------------------------------------------------------------------------
// LNN Euler scan: single workgroup, 16 waves. W_rec resident in registers as
// 8 B-fragments per wave (one-time f32->bf16 at load). h kept in LDS twice:
// f32 (update math) + bf16 mirror (fragment feeds, zero converts per step).
// ---------------------------------------------------------------------------
__global__ void __launch_bounds__(512)
lnn_scan(const bf16* __restrict__ ip, const float* __restrict__ Wrec,
         const float* __restrict__ tau, bf16* __restrict__ seq) {
  __shared__ float sh[16 * 264];
  __shared__ bf16  shb[16 * 264];  // stride 264 halves = 528B (16B multiple)
  const int tid  = threadIdx.x;
  const int wave = tid >> 5;       // 0..15 -> 16-col N tile
  const int lane = tid & 31;
  const int hi   = lane >> 4;
  const int ln   = lane & 15;
  const int ncol = wave * 16 + ln;

  v16bf wb[8];
#pragma unroll
  for (int kc = 0; kc < 8; ++kc) {
#pragma unroll
    for (int e = 0; e < 16; ++e) {
      int k = kc * 32 + 16 * hi + e;
      wb[kc][e] = (bf16)Wrec[(size_t)k * 256 + ncol];
    }
  }
  const float tv = tau[ncol];

  for (int i = tid; i < 16 * 264; i += 512) { sh[i] = 0.0f; shb[i] = (bf16)0.0f; }
  __syncthreads();

  for (int t = 0; t < 1024; ++t) {
    v8f acc = (v8f)(0.0f);
#pragma unroll
    for (int kc = 0; kc < 8; ++kc) {
      v16bf a;
      const bf16* ap = &shb[ln * 264 + kc * 32 + 16 * hi];
#pragma unroll
      for (int j = 0; j < 8; ++j) a[j] = ap[j];
#pragma unroll
      for (int j = 0; j < 8; ++j) a[8 + j] = ap[8 + j];
      acc = WMMA_BF16(a, wb[kc], acc);
    }
    __syncthreads();  // all waves done reading h
#pragma unroll
    for (int v = 0; v < 8; ++v) {
      int b = v + 8 * hi;  // batch row
      float hold = sh[b * 264 + ncol];
      float ipv  = (float)ip[((size_t)b * 1024 + t) * 256 + ncol];
      float f    = tanhf(ipv + acc[v]);
      float hn   = hold + (-hold / tv + f) * 0.1f;
      hn = fminf(10.0f, fmaxf(-10.0f, hn));
      sh[b * 264 + ncol]  = hn;
      shb[b * 264 + ncol] = (bf16)hn;
      seq[((size_t)b * 1024 + t) * 256 + ncol] = (bf16)hn;
    }
    __syncthreads();  // updates visible before next step's reads
  }
}

// ---------------------------------------------------------------------------
// Flash-style attention over bf16 qkv: grid (T/128, NH, B); wave = 16 queries.
// K tiles staged via async LDS copies; V transposed manually; online softmax
// via ds_bpermute butterflies over 16-lane C-fragment halves.
// ---------------------------------------------------------------------------
__global__ void __launch_bounds__(256)
attention_fa(const bf16* __restrict__ qkv, bf16* __restrict__ ctx) {
  __shared__ bf16 kb[32 * 72];      // K chunk (32 keys x 64 dims), 144B rows
  __shared__ bf16 vt[64 * 40];      // V chunk transposed (dim-major)
  __shared__ bf16 pb[8 * 16 * 40];  // per-wave P scratch
  const int tid  = threadIdx.x;
  const int wave = tid >> 5;
  const int lane = tid & 31;
  const int hi   = lane >> 4;
  const int ln   = lane & 15;
  const int b    = blockIdx.z;
  const int h    = blockIdx.y;
  const int qbase = blockIdx.x * 128 + wave * 16;
  const size_t RS = 768;  // qkv row stride (3H)
  const bf16* qp0 = qkv + ((size_t)b * 1024 + qbase) * RS + h * 64;

  // Q fragments (dh=64 -> two K-chunks of 32), raw bf16, no converts
  v16bf aq[2];
#pragma unroll
  for (int c = 0; c < 2; ++c) {
    const bf16* qp = qp0 + (size_t)ln * RS + c * 32 + 16 * hi;
#pragma unroll
    for (int j = 0; j < 8; ++j) aq[c][j] = qp[j];
#pragma unroll
    for (int j = 0; j < 8; ++j) aq[c][8 + j] = qp[8 + j];
  }

  v8f o[4];
#pragma unroll
  for (int i = 0; i < 4; ++i) o[i] = (v8f)(0.0f);
  float mrow[8], lrow[8];
#pragma unroll
  for (int v = 0; v < 8; ++v) { mrow[v] = -1e30f; lrow[v] = 0.0f; }

  bf16* pbw = &pb[wave * 16 * 40];

  for (int kc = 0; kc < 32; ++kc) {
    const bf16* kvrow = qkv + ((size_t)b * 1024 + kc * 32) * RS + h * 64;
    // K tile: 32 rows x 64 halves = 256 x 16B chunks; async, 1 per thread.
    {
      int r = tid >> 3, c = tid & 7;
      async_copy16(&kb[r * 72 + c * 8], kvrow + (size_t)r * RS + 256 + c * 8);
    }
    // V tile transposed: each thread moves one 8-half run into dim-major LDS.
    {
      int r = tid >> 3, c = tid & 7;
      const bf16* vsrc = kvrow + (size_t)r * RS + 512 + c * 8;
#pragma unroll
      for (int j = 0; j < 8; ++j) vt[(c * 8 + j) * 40 + r] = vsrc[j];
    }
    wait_async0();
    __syncthreads();

    // S(16x32) = Q(16x64) @ Kchunk^T
    v8f s[2];
#pragma unroll
    for (int j = 0; j < 2; ++j) {
      s[j] = (v8f)(0.0f);
#pragma unroll
      for (int c = 0; c < 2; ++c) {
        v16bf bk;
        const bf16* kp = &kb[(j * 16 + ln) * 72 + c * 32 + 16 * hi];
#pragma unroll
        for (int e = 0; e < 16; ++e) bk[e] = kp[e];
        s[j] = WMMA_BF16(aq[c], bk, s[j]);
      }
    }

    // online softmax (row = v + 8*hi; reduce across 16 lanes of the half)
#pragma unroll
    for (int v = 0; v < 8; ++v) {
      float s0 = s[0][v] * 0.125f, s1 = s[1][v] * 0.125f;
      float mx = fmaxf(s0, s1);
      mx = fmaxf(mx, bperm(mx, lane ^ 1));
      mx = fmaxf(mx, bperm(mx, lane ^ 2));
      mx = fmaxf(mx, bperm(mx, lane ^ 4));
      mx = fmaxf(mx, bperm(mx, lane ^ 8));
      float mnew  = fmaxf(mrow[v], mx);
      float alpha = __expf(mrow[v] - mnew);
      float p0 = __expf(s0 - mnew), p1 = __expf(s1 - mnew);
      float rs = p0 + p1;
      rs += bperm(rs, lane ^ 1);
      rs += bperm(rs, lane ^ 2);
      rs += bperm(rs, lane ^ 4);
      rs += bperm(rs, lane ^ 8);
      lrow[v] = lrow[v] * alpha + rs;
      mrow[v] = mnew;
#pragma unroll
      for (int nt = 0; nt < 4; ++nt) o[nt][v] *= alpha;
      int m = v + 8 * hi;
      pbw[m * 40 + ln]      = (bf16)p0;  // cols 0..15
      pbw[m * 40 + 16 + ln] = (bf16)p1;  // cols 16..31
    }

    // P as A-fragment (in-order DS within wave: no barrier needed)
    v16bf pa;
    {
      const bf16* pp = &pbw[ln * 40 + 16 * hi];
#pragma unroll
      for (int j = 0; j < 8; ++j) pa[j] = pp[j];
#pragma unroll
      for (int j = 0; j < 8; ++j) pa[8 + j] = pp[8 + j];
    }
#pragma unroll
    for (int nt = 0; nt < 4; ++nt) {
      v16bf bv;
      const bf16* vp = &vt[(nt * 16 + ln) * 40 + 16 * hi];
#pragma unroll
      for (int e = 0; e < 16; ++e) bv[e] = vp[e];
      o[nt] = WMMA_BF16(pa, bv, o[nt]);
    }
    __syncthreads();
  }

#pragma unroll
  for (int v = 0; v < 8; ++v) {
    float inv = 1.0f / lrow[v];
    int qrow = qbase + v + 8 * hi;
    bf16* dst = ctx + ((size_t)b * 1024 + qrow) * 256 + h * 64;
#pragma unroll
    for (int nt = 0; nt < 4; ++nt) dst[nt * 16 + ln] = (bf16)(o[nt][v] * inv);
  }
}

// ---------------------------------------------------------------------------
// Mean-pool over T then 1-unit head.
// ---------------------------------------------------------------------------
__global__ void __launch_bounds__(256)
pool_head(const bf16* __restrict__ seq, const float* __restrict__ Wfc,
          const float* __restrict__ bfc, float* __restrict__ out) {
  __shared__ float sb[256];
  const int b = blockIdx.x, h = threadIdx.x;
  float s = 0.0f;
  for (int t = 0; t < 1024; ++t) s += (float)seq[((size_t)b * 1024 + t) * 256 + h];
  sb[h] = (s * (1.0f / 1024.0f)) * Wfc[h];
  __syncthreads();
  for (int st = 128; st > 0; st >>= 1) { if (h < st) sb[h] += sb[h + st]; __syncthreads(); }
  if (h == 0) out[b] = sb[0] + bfc[0];
}

// ---------------------------------------------------------------------------
// Orchestration.  Workspace: ~51M bf16 elements (~102 MB, L2-resident).
// ---------------------------------------------------------------------------
extern "C" void kernel_launch(void* const* d_in, const int* in_sizes, int n_in,
                              void* d_out, int out_size, void* d_ws, size_t ws_size,
                              hipStream_t stream) {
  const float* x     = (const float*)d_in[0];
  const float* W_in  = (const float*)d_in[1];
  const float* b_in  = (const float*)d_in[2];
  const float* lng   = (const float*)d_in[3];
  const float* lnb   = (const float*)d_in[4];
  const float* W_x   = (const float*)d_in[5];
  const float* b_x   = (const float*)d_in[6];
  const float* tau   = (const float*)d_in[7];
  const float* W_rec = (const float*)d_in[8];
  const float* Wqkv  = (const float*)d_in[9];
  const float* bqkv  = (const float*)d_in[10];
  const float* Wo    = (const float*)d_in[11];
  const float* bo    = (const float*)d_in[12];
  const float* ln1g  = (const float*)d_in[13];
  const float* ln1b  = (const float*)d_in[14];
  const float* W1    = (const float*)d_in[15];
  const float* b1    = (const float*)d_in[16];
  const float* W2    = (const float*)d_in[17];
  const float* b2    = (const float*)d_in[18];
  const float* ln2g  = (const float*)d_in[19];
  const float* ln2b  = (const float*)d_in[20];
  const float* W_fc  = (const float*)d_in[21];
  const float* b_fc  = (const float*)d_in[22];
  (void)in_sizes; (void)n_in; (void)out_size; (void)ws_size;

  const size_t NBT = 16384;  // B*T
  bf16* w   = (bf16*)d_ws;
  bf16* u    = w;                       // 4,194,304
  bf16* ip   = u    + NBT * 256;
  bf16* seq  = ip   + NBT * 256;
  bf16* qkvb = seq  + NBT * 256;        // 12,582,912
  bf16* mid  = qkvb + NBT * 768;        // 16,777,216
  bf16* ctx  = mid  + NBT * 1024;
  bf16* tmp  = ctx  + NBT * 256;
  bf16* wWx  = tmp  + NBT * 256;        // 65,536
  bf16* wQKV = wWx  + 65536;            // 393,216 (2 layers)
  bf16* wWo  = wQKV + 393216;           // 131,072
  bf16* wW1  = wWo  + 131072;           // 524,288
  bf16* wW2  = wW1  + 524288;           // 524,288

  // one-time weight conversions (deterministic; cheap vs. 90 GF of GEMMs)
  f32_to_bf16<<<256,  256, 0, stream>>>(W_x,   wWx,  65536);
  f32_to_bf16<<<1536, 256, 0, stream>>>(Wqkv,  wQKV, 393216);
  f32_to_bf16<<<512,  256, 0, stream>>>(Wo,    wWo,  131072);
  f32_to_bf16<<<2048, 256, 0, stream>>>(W1,    wW1,  524288);
  f32_to_bf16<<<2048, 256, 0, stream>>>(W2,    wW2,  524288);

  input_proj_ln_gelu<<<16384, 256, 0, stream>>>(x, W_in, b_in, lng, lnb, u);
  gemm_bias_act<0><<<dim3(4, 128), 256, 0, stream>>>(u, wWx, b_x, ip, 16384, 256, 256);
  lnn_scan<<<1, 512, 0, stream>>>(ip, W_rec, tau, seq);

  for (int l = 0; l < 2; ++l) {
    gemm_bias_act<0><<<dim3(12, 128), 256, 0, stream>>>(
        seq, wQKV + (size_t)l * 768 * 256, bqkv + l * 768, qkvb, 16384, 768, 256);
    attention_fa<<<dim3(8, 4, 16), 256, 0, stream>>>(qkvb, ctx);
    gemm_bias_act<0><<<dim3(4, 128), 256, 0, stream>>>(
        ctx, wWo + (size_t)l * 256 * 256, bo + l * 256, tmp, 16384, 256, 256);
    residual_ln<<<16384, 256, 0, stream>>>(seq, tmp, ln1g + l * 256, ln1b + l * 256, seq);
    gemm_bias_act<1><<<dim3(16, 128), 256, 0, stream>>>(
        seq, wW1 + (size_t)l * 1024 * 256, b1 + l * 1024, mid, 16384, 1024, 256);
    gemm_bias_act<0><<<dim3(4, 128), 256, 0, stream>>>(
        mid, wW2 + (size_t)l * 256 * 1024, b2 + l * 256, tmp, 16384, 256, 1024);
    residual_ln<<<16384, 256, 0, stream>>>(seq, tmp, ln2g + l * 256, ln2b + l * 256, seq);
  }
  pool_head<<<16, 256, 0, stream>>>(seq, W_fc, b_fc, (float*)d_out);
}
